// TriIn_20194936225859
// MI455X (gfx1250) — matrix operand order repaired and checked by reading, not compiled, not run on hardware
//
#include <hip/hip_runtime.h>
#include <cstdint>
#include <cstddef>

#define Ldim 512
#define ZDim 128
#define CDim 128

typedef __attribute__((ext_vector_type(16))) __bf16 v16bf;
typedef __attribute__((ext_vector_type(8)))  __bf16 v8bf;
typedef __attribute__((ext_vector_type(8)))  float  v8f;

// workspace layout (bytes)
#define WS_AW   ((size_t)0)
#define WS_AGW  ((size_t)32768)
#define WS_GW   ((size_t)65536)
#define WS_OW   ((size_t)98304)
#define WS_A    ((size_t)131072)                                   // bf16 [C][i=L][l=L]
#define WS_G    (WS_A + (size_t)CDim * Ldim * Ldim * 2)            // bf16 [p=L*L][C]

__device__ __forceinline__ float sigmoidf_(float x) { return 1.f / (1.f + __expf(-x)); }

// ---------------------------------------------------------------------------
// k0: cast the four 128x128 f32 weight matrices to bf16 (stay L2-resident)
// ---------------------------------------------------------------------------
__global__ void k0_cast_weights(const float* __restrict__ aW, const float* __restrict__ agW,
                                const float* __restrict__ gW, const float* __restrict__ oW,
                                __bf16* __restrict__ dA, __bf16* __restrict__ dAG,
                                __bf16* __restrict__ dG, __bf16* __restrict__ dO) {
  int t = blockIdx.x * blockDim.x + threadIdx.x;     // 0..65535
  int which = t >> 14, idx = t & 16383;
  const float* s = (which == 0) ? aW : (which == 1) ? agW : (which == 2) ? gW : oW;
  __bf16* d = (which == 0) ? dA : (which == 1) ? dAG : (which == 2) ? dG : dO;
  d[idx] = (__bf16)s[idx];
}

// ---------------------------------------------------------------------------
// k1: LayerNorm(z) -> three WMMA GEMMs (a-linear, a-gate, out-gate) -> fuse.
// One block = 128 consecutive positions (fixed l, i0..i0+127), 8 waves.
//   a = (zn@aW^T+ab) * sigmoid(zn@agW^T+agb)   stored bf16 at ws_a[c][i][l]
//   g = sigmoid(zn@gW^T+gb)                    stored bf16 at ws_g[p][c]
// ---------------------------------------------------------------------------
__global__ void k1_ln_proj(const float* __restrict__ z,
                           const float* __restrict__ ng, const float* __restrict__ nb,
                           const __bf16* __restrict__ wA,  const float* __restrict__ ab,
                           const __bf16* __restrict__ wAG, const float* __restrict__ agb,
                           const __bf16* __restrict__ wG,  const float* __restrict__ gb,
                           __bf16* __restrict__ outA, __bf16* __restrict__ outG) {
  __shared__ __bf16 zn[128 * ZDim];                  // 32 KB normalized activations

  const int tid = threadIdx.x;                       // 0..255
  const size_t p0 = (size_t)blockIdx.x * 128;
  const int l_row = (int)(p0 / Ldim);
  const int i0    = (int)(p0 % Ldim);

  // LayerNorm: threads 0..127 each own one row, vectorized float4 (b128) loads
  if (tid < 128) {
    const float4* row4 = (const float4*)(z + (p0 + tid) * ZDim);
    float s = 0.f, s2 = 0.f;
#pragma unroll 8
    for (int k4 = 0; k4 < ZDim / 4; ++k4) {
      float4 x = row4[k4];
      s  += x.x + x.y + x.z + x.w;
      s2 += x.x * x.x + x.y * x.y + x.z * x.z + x.w * x.w;
    }
    float mu = s * (1.f / ZDim);
    float var = s2 * (1.f / ZDim) - mu * mu;
    float rs = rsqrtf(var + 1e-5f);
    __bf16* orow = zn + tid * ZDim;
#pragma unroll 8
    for (int k4 = 0; k4 < ZDim / 4; ++k4) {
      float4 x = row4[k4];                            // L0/L1-hot re-read
      orow[k4 * 4 + 0] = (__bf16)(((x.x - mu) * rs) * ng[k4 * 4 + 0] + nb[k4 * 4 + 0]);
      orow[k4 * 4 + 1] = (__bf16)(((x.y - mu) * rs) * ng[k4 * 4 + 1] + nb[k4 * 4 + 1]);
      orow[k4 * 4 + 2] = (__bf16)(((x.z - mu) * rs) * ng[k4 * 4 + 2] + nb[k4 * 4 + 2]);
      orow[k4 * 4 + 3] = (__bf16)(((x.w - mu) * rs) * ng[k4 * 4 + 3] + nb[k4 * 4 + 3]);
    }
  }
  __syncthreads();

  const int wave = tid >> 5;
  const int lane = tid & 31;
  const int n    = lane & 15;
  const int hi   = lane >> 4;
  const int kbA  = hi ? 16 : 0;      // A fragment K base (contiguous k = 16h + s, matches B)
  const int kbB  = hi ? 16 : 0;      // B fragment K base (contiguous k = 16h + s, matches A)
  const int mhi  = hi ? 8 : 0;       // D fragment M offset
  const int wbase = wave * 16;       // this wave's 16-row M tile

  const v8f vzero = {0.f, 0.f, 0.f, 0.f, 0.f, 0.f, 0.f, 0.f};

  // A fragments (invariant across output-channel tiles): hoist all 4 K-steps
  v16bf afr[4];
#pragma unroll
  for (int ks = 0; ks < 4; ++ks) {
    const __bf16* pa = zn + (wbase + n) * ZDim + ks * 32 + kbA;
    ((v8bf*)&afr[ks])[0] = *(const v8bf*)(pa);
    ((v8bf*)&afr[ks])[1] = *(const v8bf*)(pa + 8);
  }

  for (int nt = 0; nt < 8; ++nt) {   // 8 output-channel tiles of 16
    v8f accA = vzero, accAG = vzero, accG = vzero;
#pragma unroll
    for (int ks = 0; ks < 4; ++ks) { // K = 128 in 4 steps of 32
      const int wrow = nt * 16 + n;
      v16bf bA  = *(const v16bf*)(wA  + wrow * ZDim + ks * 32 + kbB);
      v16bf bAG = *(const v16bf*)(wAG + wrow * ZDim + ks * 32 + kbB);
      v16bf bG  = *(const v16bf*)(wG  + wrow * ZDim + ks * 32 + kbB);
      accA  = __builtin_amdgcn_wmma_f32_16x16x32_bf16(false, afr[ks], false, bA,  (short)0, accA,  false, false);
      accAG = __builtin_amdgcn_wmma_f32_16x16x32_bf16(false, afr[ks], false, bAG, (short)0, accAG, false, false);
      accG  = __builtin_amdgcn_wmma_f32_16x16x32_bf16(false, afr[ks], false, bG,  (short)0, accG,  false, false);
    }
    const int c = nt * 16 + n;
    const float biasA = ab[c], biasAG = agb[c], biasG = gb[c];
#pragma unroll
    for (int v = 0; v < 8; ++v) {
      const int m = wbase + v + mhi;                 // row within block's 128 positions
      const float av = (accA[v] + biasA) * sigmoidf_(accAG[v] + biasAG);
      outA[((size_t)c << 18) + (size_t)(i0 + m) * Ldim + l_row] = (__bf16)av;
      outG[(p0 + m) * CDim + c] = (__bf16)sigmoidf_(accG[v] + biasG);
    }
  }
}

// ---------------------------------------------------------------------------
// k2: fused einsum + channel-LayerNorm + oW projection + gating.
// One block = one 16x16 (i,j) output tile; 8 waves x 16 channels each.
// Einsum: O[i,j,c] = sum_l A_c[i][l]*A_c[j][l]; K contiguous in ws_a (L2-hot).
// ---------------------------------------------------------------------------
__global__ void k2_einsum_out(const __bf16* __restrict__ A,   // [C][i=L][l=L]
                              const __bf16* __restrict__ G,   // [p][C] gate
                              const float* __restrict__ og, const float* __restrict__ obn,
                              const __bf16* __restrict__ wO, const float* __restrict__ ob,
                              float* __restrict__ out) {
  __shared__ __bf16 Ol[256 * CDim];                  // 64 KB, normalized in place

  const int tid  = threadIdx.x;
  const int wave = tid >> 5;
  const int lane = tid & 31;
  const int n    = lane & 15;
  const int hi   = lane >> 4;
  const int kbA  = hi ? 16 : 0;
  const int kbB  = hi ? 16 : 0;
  const int mhi  = hi ? 8 : 0;

  const int i0 = blockIdx.x * 16;
  const int j0 = blockIdx.y * 16;
  const int cw = wave * 16;                          // this wave's channel group

  const v8f vzero = {0.f, 0.f, 0.f, 0.f, 0.f, 0.f, 0.f, 0.f};
  v8f acc[16];
#pragma unroll
  for (int c = 0; c < 16; ++c) acc[c] = vzero;

  const __bf16* baseI = A + (size_t)(i0 + n) * Ldim; // A-frag row (M = lane&15)
  const __bf16* baseJ = A + (size_t)(j0 + n) * Ldim; // B-frag row (N = lane&15)

  for (int l0 = 0; l0 < Ldim; l0 += 32) {            // 16 K-steps of 32
#pragma unroll
    for (int c = 0; c < 16; ++c) {
      const size_t coff = ((size_t)(cw + c)) << 18;  // channel plane (L*L)
      v16bf af;
      const __bf16* pa = baseI + coff + l0 + kbA;
      ((v8bf*)&af)[0] = *(const v8bf*)(pa);
      ((v8bf*)&af)[1] = *(const v8bf*)(pa + 8);
      v16bf bf = *(const v16bf*)(baseJ + coff + l0 + kbB);
      acc[c] = __builtin_amdgcn_wmma_f32_16x16x32_bf16(false, af, false, bf, (short)0, acc[c], false, false);
    }
  }

  // scatter tile into LDS as [pos = mi*16+nj][c]
#pragma unroll
  for (int c = 0; c < 16; ++c)
#pragma unroll
    for (int v = 0; v < 8; ++v) {
      const int mi = v + mhi;
      Ol[(mi * 16 + n) * CDim + cw + c] = (__bf16)acc[c][v];
    }
  __syncthreads();

  // LayerNorm over the 128 channels, in place (thread t owns position t),
  // vectorized as 16-byte LDS accesses.
  {
    v8bf* row8 = (v8bf*)(Ol + tid * CDim);
    float s = 0.f, s2 = 0.f;
#pragma unroll
    for (int kk = 0; kk < CDim / 8; ++kk) {
      v8bf ch = row8[kk];
#pragma unroll
      for (int e = 0; e < 8; ++e) { float x = (float)ch[e]; s += x; s2 += x * x; }
    }
    float mu = s * (1.f / CDim);
    float var = s2 * (1.f / CDim) - mu * mu;
    float rs = rsqrtf(var + 1e-5f);
#pragma unroll
    for (int kk = 0; kk < CDim / 8; ++kk) {
      v8bf ch = row8[kk];
      v8bf o;
#pragma unroll
      for (int e = 0; e < 8; ++e) {
        int k = kk * 8 + e;
        o[e] = (__bf16)((((float)ch[e] - mu) * rs) * og[k] + obn[k]);
      }
      row8[kk] = o;
    }
  }
  __syncthreads();

  // projection through oW^T + gate; each wave does 2 row-tiles of 16 positions
#pragma unroll
  for (int t = 0; t < 2; ++t) {
    const int mt = wave * 2 + t;                     // i-row within tile
    const __bf16* arow = Ol + (mt * 16 + n) * CDim;  // A-frag row (M = lane&15)

    v16bf afr[4];                                    // invariant across nt
#pragma unroll
    for (int ks = 0; ks < 4; ++ks) {
      const __bf16* pa = arow + ks * 32 + kbA;
      ((v8bf*)&afr[ks])[0] = *(const v8bf*)(pa);
      ((v8bf*)&afr[ks])[1] = *(const v8bf*)(pa + 8);
    }

#pragma unroll
    for (int nt = 0; nt < 8; ++nt) {
      v8f pacc = vzero;
#pragma unroll
      for (int ks = 0; ks < 4; ++ks) {
        v16bf bf = *(const v16bf*)(wO + (nt * 16 + n) * CDim + ks * 32 + kbB);
        pacc = __builtin_amdgcn_wmma_f32_16x16x32_bf16(false, afr[ks], false, bf, (short)0, pacc, false, false);
      }
      const int zd = nt * 16 + n;
      const float bias = ob[zd];
#pragma unroll
      for (int v = 0; v < 8; ++v) {
        const int nj = v + mhi;
        const size_t p = (size_t)(i0 + mt) * Ldim + (j0 + nj);
        out[p * ZDim + zd] = (pacc[v] + bias) * (float)G[p * CDim + zd];
      }
    }
  }
}

// ---------------------------------------------------------------------------
extern "C" void kernel_launch(void* const* d_in, const int* in_sizes, int n_in,
                              void* d_out, int out_size, void* d_ws, size_t ws_size,
                              hipStream_t stream) {
  const float* z   = (const float*)d_in[0];
  const float* ng  = (const float*)d_in[1];
  const float* nb  = (const float*)d_in[2];
  const float* aW  = (const float*)d_in[3];
  const float* ab  = (const float*)d_in[4];
  const float* agW = (const float*)d_in[5];
  const float* agb = (const float*)d_in[6];
  const float* og  = (const float*)d_in[7];
  const float* obn = (const float*)d_in[8];
  const float* oW  = (const float*)d_in[9];
  const float* ob  = (const float*)d_in[10];
  const float* gW  = (const float*)d_in[11];
  const float* gb  = (const float*)d_in[12];
  float* out = (float*)d_out;

  char* ws = (char*)d_ws;
  __bf16* wsAW  = (__bf16*)(ws + WS_AW);
  __bf16* wsAGW = (__bf16*)(ws + WS_AGW);
  __bf16* wsGW  = (__bf16*)(ws + WS_GW);
  __bf16* wsOW  = (__bf16*)(ws + WS_OW);
  __bf16* wsA   = (__bf16*)(ws + WS_A);
  __bf16* wsG   = (__bf16*)(ws + WS_G);

  k0_cast_weights<<<dim3(256), dim3(256), 0, stream>>>(aW, agW, gW, oW,
                                                       wsAW, wsAGW, wsGW, wsOW);
  k1_ln_proj<<<dim3((Ldim * Ldim) / 128), dim3(256), 0, stream>>>(
      z, ng, nb, wsAW, ab, wsAGW, agb, wsGW, gb, wsA, wsG);
  k2_einsum_out<<<dim3(Ldim / 16, Ldim / 16), dim3(256), 0, stream>>>(
      wsA, wsG, og, obn, wsOW, ob, out);
}
